// FP8GEMMS_77403900609092
// MI455X (gfx1250) — 8-bit matrix operand order repaired and checked by reading, compiled, not run on hardware
//
#include <hip/hip_runtime.h>
#include <hip/hip_bf16.h>

typedef __attribute__((ext_vector_type(16))) int   v16i;
typedef __attribute__((ext_vector_type(8)))  float v8f;
typedef __attribute__((ext_vector_type(4)))  int   v4i;
typedef __attribute__((ext_vector_type(2)))  int   v2i;
typedef __attribute__((ext_vector_type(4)))  float v4f;

#define RR 4
#define MM 4096
#define NN 4096
#define KK 512
#define KT (RR * KK)

__device__ inline unsigned char quant_e4m3(float x) {
  unsigned u = __float_as_uint(x);
  unsigned sgn = (u >> 24) & 0x80u;
  float a = fabsf(x);
  if (!(a >= 0.015625f)) {
    int mant = (int)rintf(a * 512.0f);
    if (mant > 7) return (unsigned char)(sgn | 0x08u);
    return (unsigned char)(sgn | (unsigned)mant);
  }
  if (a >= 448.0f) return (unsigned char)(sgn | 0x7Eu);
  int e;
  float m = frexpf(a, &e);
  int mant = (int)rintf(m * 16.0f) - 8;
  int E = e - 1 + 7;
  if (mant == 8) { mant = 0; ++E; }
  if (E > 15)    { E = 15; mant = 6; }
  return (unsigned char)(sgn | ((unsigned)E << 3) | (unsigned)mant);
}

__global__ void __launch_bounds__(256)
fp8_quant_kernel(const float* __restrict__ x1, const float* __restrict__ x2,
                 const float* __restrict__ s1, const float* __restrict__ s2,
                 unsigned char* __restrict__ Aq, unsigned char* __restrict__ Bq) {
  const int mat = blockIdx.y;
  const float* __restrict__ x = mat ? x2 : x1;
  const float scale = mat ? s2[0] : s1[0];
  unsigned char* __restrict__ outp = mat ? Bq : Aq;

  size_t o = ((size_t)blockIdx.x * blockDim.x + threadIdx.x) * 4;
  if (o >= (size_t)MM * KT) return;
  size_t m   = o / KT;
  size_t rem = o - m * KT;
  size_t r   = rem / KK;
  size_t k   = rem - r * KK;
  const v4f src = *(const v4f*)(x + (r * (size_t)MM + m) * KK + k);

  unsigned int packed = 0;
#pragma unroll
  for (int i = 0; i < 4; ++i)
    packed |= (unsigned int)quant_e4m3(src[i] * scale) << (8 * i);
  *(unsigned int*)(outp + o) = packed;
}

union AF { int   i[16]; v16i v; };
union CF { float f[8];  v8f  v; };

__global__ void __launch_bounds__(256)
fp8_gemm_kernel(const unsigned char* __restrict__ Aq,
                const unsigned char* __restrict__ Bq,
                const float* __restrict__ si1, const float* __restrict__ si2,
                __hip_bfloat16* __restrict__ C) {
  const int lane     = (int)(threadIdx.x & 31u);
  const int wave     = (int)(threadIdx.x >> 5);
  const int laneLow  = lane & 15;
  const int laneHalf = lane >> 4;
  const int waveM    = wave >> 2;
  const int waveN    = wave & 3;
  const int mbase = (int)blockIdx.y * 128 + waveM * 64;
  const int nbase = (int)blockIdx.x * 128 + waveN * 32;

  CF acc[4][2];
#pragma unroll
  for (int mt = 0; mt < 4; ++mt)
#pragma unroll
    for (int nt = 0; nt < 2; ++nt)
      acc[mt][nt].v = (v8f){0.f, 0.f, 0.f, 0.f, 0.f, 0.f, 0.f, 0.f};

  for (int k0 = 0; k0 < KT; k0 += 128) {
    AF a[4];
    AF b[2];
#pragma unroll
    for (int mt = 0; mt < 4; ++mt) {
      const unsigned char* arow =
          Aq + (size_t)(mbase + mt * 16 + laneLow) * KT + k0 + laneHalf * 16;
      __builtin_prefetch(arow + 128, 0, 1);
#pragma unroll
      for (int j = 0; j < 8; ++j) {
        v2i c = *(const v2i*)(arow + (j >> 1) * 32 + (j & 1) * 8);
        a[mt].i[2 * j]     = c[0];
        a[mt].i[2 * j + 1] = c[1];
      }
    }
#pragma unroll
    for (int nt = 0; nt < 2; ++nt) {
      const unsigned char* brow =
          Bq + (size_t)(nbase + nt * 16 + laneLow) * KT + k0 + laneHalf * 16;
      __builtin_prefetch(brow + 128, 0, 1);
#pragma unroll
      for (int j = 0; j < 4; ++j) {
        v4i c = *(const v4i*)(brow + j * 32);
        b[nt].i[4 * j]     = c[0];
        b[nt].i[4 * j + 1] = c[1];
        b[nt].i[4 * j + 2] = c[2];
        b[nt].i[4 * j + 3] = c[3];
      }
    }
#pragma unroll
    for (int mt = 0; mt < 4; ++mt)
#pragma unroll
      for (int nt = 0; nt < 2; ++nt)
        acc[mt][nt].v = __builtin_amdgcn_wmma_f32_16x16x128_fp8_fp8(
            a[mt].v, b[nt].v, (short)0, acc[mt][nt].v, false, false);
  }

  const float sc = si1[0] * si2[0];
#pragma unroll
  for (int mt = 0; mt < 4; ++mt) {
#pragma unroll
    for (int nt = 0; nt < 2; ++nt) {
      const int col = nbase + nt * 16 + laneLow;
#pragma unroll
      for (int v = 0; v < 8; ++v) {
        const int row = mbase + mt * 16 + laneHalf * 8 + v;
        C[(size_t)row * NN + col] = __float2bfloat16(acc[mt][nt].f[v] * sc);
      }
    }
  }
}

extern "C" void kernel_launch(void* const* d_in, const int* in_sizes, int n_in,
                              void* d_out, int out_size, void* d_ws, size_t ws_size,
                              hipStream_t stream) {
  (void)in_sizes; (void)n_in; (void)out_size; (void)ws_size;
  const float* x1  = (const float*)d_in[0];
  const float* x2  = (const float*)d_in[1];
  const float* s1  = (const float*)d_in[2];
  const float* s2  = (const float*)d_in[3];
  const float* si1 = (const float*)d_in[4];
  const float* si2 = (const float*)d_in[5];
  __hip_bfloat16* out = (__hip_bfloat16*)d_out;

  unsigned char* Aq = (unsigned char*)d_ws;
  unsigned char* Bq = Aq + (size_t)MM * KT;

  const size_t bytes_per_mat = (size_t)MM * KT;
  dim3 qgrid((unsigned)((bytes_per_mat / 4 + 255) / 256), 2, 1);
  fp8_quant_kernel<<<qgrid, 256, 0, stream>>>(x1, x2, s1, s2, Aq, Bq);

  dim3 ggrid(NN / 128, MM / 128, 1);
  fp8_gemm_kernel<<<ggrid, 256, 0, stream>>>(Aq, Bq, si1, si2, out);
}
